// NeuralBlendshapes_24833500905483
// MI455X (gfx1250) — matrix operand order repaired and checked by reading, not compiled, not run on hardware
//
#include <hip/hip_runtime.h>
#include <hip/hip_bf16.h>

typedef __attribute__((ext_vector_type(16))) _Float16 v16h;
typedef __attribute__((ext_vector_type(8)))  _Float16 v8h;
typedef __attribute__((ext_vector_type(8)))  float    v8f;

// ---- packed f16 weight layout (element offsets, halves) ----
#define OFF_W0A 0        // [256][96]  (ew0[:, :78] zero-padded to 96)
#define OFF_W1  24576    // [256][256]
#define OFF_W2  90112    // [128][256]
#define OFF_W3  122880   // [128][128]
#define OFF_W4  139264   // [128][128]
#define OFF_W5  155648   // [16][128]  (ew5 rows 3..15 zero)
#define W16_TOTAL 157696

__device__ __forceinline__ float silu_f(float x){ return x / (1.0f + __expf(-x)); }
__device__ __forceinline__ float sigm_f(float x){ return 1.0f / (1.0f + __expf(-x)); }

// =====================================================================
// K0a: pack expression-MLP weights to f16, [N][K] row-major, padded
// =====================================================================
__global__ void pack_weights(const float* __restrict__ ew0, const float* __restrict__ ew1,
                             const float* __restrict__ ew2, const float* __restrict__ ew3,
                             const float* __restrict__ ew4, const float* __restrict__ ew5,
                             _Float16* __restrict__ w16)
{
    int i = blockIdx.x * 256 + threadIdx.x;
    if (i >= W16_TOTAL) return;
    if (i < OFF_W1) {                       // W0a: 256 x 96 from ew0 (256 x 131)
        int j = i - OFF_W0A, n = j / 96, k = j % 96;
        w16[i] = (_Float16)((k < 78) ? ew0[n * 131 + k] : 0.0f);
    } else if (i < OFF_W2) {
        w16[i] = (_Float16)ew1[i - OFF_W1];
    } else if (i < OFF_W3) {
        w16[i] = (_Float16)ew2[i - OFF_W2];
    } else if (i < OFF_W4) {
        w16[i] = (_Float16)ew3[i - OFF_W3];
    } else if (i < OFF_W5) {
        w16[i] = (_Float16)ew4[i - OFF_W4];
    } else {                                // W5: 16 x 128 from ew5 (3 x 128)
        int j = i - OFF_W5, n = j / 128, k = j % 128;
        w16[i] = (_Float16)((n < 3) ? ew5[n * 128 + k] : 0.0f);
    }
}

// =====================================================================
// K0b: fold constant feature columns into per-batch biases
// ebias[b][n] = eb0[n] + sum_j ew0[n][78+j] * features[b][j]        (j<53)
// pbias[b][n] = pb0[n] + sum_j pw0[n][21+j] * features[b][53+j]     (j<6)
// =====================================================================
__global__ void bias_kernel(const float* __restrict__ features,
                            const float* __restrict__ ew0, const float* __restrict__ eb0,
                            const float* __restrict__ pw0, const float* __restrict__ pb0,
                            float* __restrict__ ebias, float* __restrict__ pbias)
{
    int b = blockIdx.x, t = threadIdx.x;
    __shared__ float f[62];
    if (t < 62) f[t] = features[b * 62 + t];
    __syncthreads();
    if (t < 256) {
        float s = eb0[t];
        #pragma unroll 1
        for (int j = 0; j < 53; ++j) s += ew0[t * 131 + 78 + j] * f[j];
        ebias[b * 256 + t] = s;
    }
    if (t < 64) {
        float s = pb0[t];
        #pragma unroll
        for (int j = 0; j < 6; ++j) s += pw0[t * 27 + 21 + j] * f[53 + j];
        pbias[b * 64 + t] = s;
    }
}

// =====================================================================
// per-thread dense helper (fully unrolled, activations stay in VGPRs)
// =====================================================================
template<int IN, int OUT, int STRIDE, bool ACT>
__device__ __forceinline__ void dense(const float* __restrict__ W, const float* __restrict__ bvec,
                                      const float* in, float* out)
{
    #pragma unroll
    for (int n = 0; n < OUT; ++n) {
        float s = bvec[n];
        #pragma unroll
        for (int k = 0; k < IN; ++k) s += W[n * STRIDE + k] * in[k];
        out[n] = ACT ? silu_f(s) : s;
    }
}

// =====================================================================
// K1: template MLP per vertex (21 -> 64 -> 32 -> 32 -> 3), scalar
// =====================================================================
__global__ void tmpl_kernel(const float* __restrict__ verts,
                            const float* __restrict__ tw0, const float* __restrict__ tb0,
                            const float* __restrict__ tw1, const float* __restrict__ tb1,
                            const float* __restrict__ tw2, const float* __restrict__ tb2,
                            const float* __restrict__ tw3, const float* __restrict__ tb3,
                            float* __restrict__ tmpl, int V)
{
    int v = blockIdx.x * 256 + threadIdx.x;
    if (v >= V) return;
    float x0 = verts[v * 6 + 0], x1 = verts[v * 6 + 1], x2 = verts[v * 6 + 2];
    float e[21];
    e[0] = x0; e[1] = x1; e[2] = x2;
    #pragma unroll
    for (int n = 0; n < 3; ++n) {
        float s = (float)(1 << n);
        e[3 + 6 * n + 0] = __sinf(x0 * s); e[3 + 6 * n + 1] = __sinf(x1 * s); e[3 + 6 * n + 2] = __sinf(x2 * s);
        e[3 + 6 * n + 3] = __cosf(x0 * s); e[3 + 6 * n + 4] = __cosf(x1 * s); e[3 + 6 * n + 5] = __cosf(x2 * s);
    }
    float h0[64], h1[32], h2[32], o[3];
    dense<21, 64, 21, true >(tw0, tb0, e,  h0);
    dense<64, 32, 64, true >(tw1, tb1, h0, h1);
    dense<32, 32, 32, true >(tw2, tb2, h1, h2);
    dense<32,  3, 32, false>(tw3, tb3, h2, o);
    tmpl[v * 3 + 0] = o[0]; tmpl[v * 3 + 1] = o[1]; tmpl[v * 3 + 2] = o[2];
}

// =====================================================================
// K2: expression MLP with WMMA f16 (K-dim 96->256->256->128->128->128->16)
// =====================================================================
__device__ __forceinline__ void stage_w(const _Float16* __restrict__ g, _Float16* lds,
                                        int nhalf, int tid)
{
    __syncthreads();                          // everyone done with previous weights
    const v8h* src = (const v8h*)g;
    v8h* dst = (v8h*)lds;
    int n8 = nhalf >> 3;
    for (int i = tid; i < n8; i += 256) dst[i] = src[i];
    __syncthreads();
}

// positional-encoding element k of the 96-wide padded layer-0 input
__device__ __forceinline__ _Float16 enc_elem(float x0, float x1, float x2,
                                             float x3, float x4, float x5, int k)
{
    if (k >= 78) return (_Float16)0.0f;
    if (k < 6) {
        float v = (k == 0) ? x0 : (k == 1) ? x1 : (k == 2) ? x2 : (k == 3) ? x3 : (k == 4) ? x4 : x5;
        return (_Float16)v;
    }
    int t = k - 6, n = t / 12, r = t % 12;
    int d = (r < 6) ? r : r - 6;
    float xv = (d == 0) ? x0 : (d == 1) ? x1 : (d == 2) ? x2 : (d == 3) ? x3 : (d == 4) ? x4 : x5;
    float v = xv * (float)(1 << n);
    return (_Float16)((r < 6) ? __sinf(v) : __cosf(v));
}

// generic WMMA layer: A from LDS [16][K] row-major, W from LDS [N][K]
template<int K, int N, bool LAST>
__device__ __forceinline__ void mlp_layer(const _Float16* __restrict__ Asrc,
                                          const _Float16* __restrict__ Wlds,
                                          const float* __restrict__ bias,
                                          _Float16* __restrict__ actOut,
                                          float* __restrict__ gOut,
                                          int lane, int rowBase, int Vtot)
{
    constexpr int NT = N / 16, KC = K / 32;
    const int rlo = lane & 15, rhalf = lane >> 4;
    v8f acc[NT];
    #pragma unroll
    for (int nt = 0; nt < NT; ++nt)
        #pragma unroll
        for (int r = 0; r < 8; ++r) acc[nt][r] = 0.0f;

    #pragma unroll
    for (int kc = 0; kc < KC; ++kc) {
        // A tile: row=lane%16; halves: K = kc*32 + (lane/16)*8 + [0..7] and +16
        const _Float16* ap = Asrc + rlo * K + kc * 32 + rhalf * 8;
        v8h alo = *(const v8h*)ap;
        v8h ahi = *(const v8h*)(ap + 16);
        v16h a;
        #pragma unroll
        for (int i = 0; i < 8; ++i) { a[i] = alo[i]; a[8 + i] = ahi[i]; }
        #pragma unroll
        for (int nt = 0; nt < NT; ++nt) {
            // B tile: halves: K = kc*32 + (lane/16)*8 + [0..7] and +16
            const _Float16* bp = Wlds + (nt * 16 + rlo) * K + kc * 32 + rhalf * 8;
            v8h blo = *(const v8h*)bp;
            v8h bhi = *(const v8h*)(bp + 16);
            v16h bm;
            #pragma unroll
            for (int i = 0; i < 8; ++i) { bm[i] = blo[i]; bm[8 + i] = bhi[i]; }
            acc[nt] = __builtin_amdgcn_wmma_f32_16x16x32_f16(
                false, a, false, bm, (short)0, acc[nt], false, false);
        }
    }
    // epilogue: D element (M = r + 8*(lane/16), N = n0 + lane%16)
    #pragma unroll
    for (int nt = 0; nt < NT; ++nt) {
        if constexpr (!LAST) {
            float bv = bias[nt * 16 + rlo];
            #pragma unroll
            for (int r = 0; r < 8; ++r) {
                float x = acc[nt][r] + bv;
                actOut[(r + 8 * rhalf) * N + nt * 16 + rlo] = (_Float16)silu_f(x);
            }
        } else {
            int col = nt * 16 + rlo;
            #pragma unroll
            for (int r = 0; r < 8; ++r) {
                int vv = rowBase + r + 8 * rhalf;
                if (col < 3 && vv < Vtot) gOut[vv * 3 + col] = acc[nt][r] + bias[col];
            }
        }
    }
}

// layer 0: A generated on the fly from positional encoding (K=96)
template<int N>
__device__ __forceinline__ void expr_layer0(const float* __restrict__ verts, int Vtot,
                                            const _Float16* __restrict__ Wlds,
                                            const float* __restrict__ bias,
                                            _Float16* __restrict__ actOut,
                                            int lane, int rowBase)
{
    constexpr int K = 96, NT = N / 16, KC = 3;
    const int rlo = lane & 15, rhalf = lane >> 4;
    int vv = rowBase + rlo; if (vv >= Vtot) vv = Vtot - 1;
    float x0 = verts[vv * 6 + 0], x1 = verts[vv * 6 + 1], x2 = verts[vv * 6 + 2];
    float x3 = verts[vv * 6 + 3], x4 = verts[vv * 6 + 4], x5 = verts[vv * 6 + 5];

    v8f acc[NT];
    #pragma unroll
    for (int nt = 0; nt < NT; ++nt)
        #pragma unroll
        for (int r = 0; r < 8; ++r) acc[nt][r] = 0.0f;

    #pragma unroll
    for (int kc = 0; kc < KC; ++kc) {
        v16h a;
        #pragma unroll
        for (int i = 0; i < 8; ++i) {
            a[i]     = enc_elem(x0, x1, x2, x3, x4, x5, kc * 32 + rhalf * 8 + i);
            a[8 + i] = enc_elem(x0, x1, x2, x3, x4, x5, kc * 32 + 16 + rhalf * 8 + i);
        }
        #pragma unroll
        for (int nt = 0; nt < NT; ++nt) {
            const _Float16* bp = Wlds + (nt * 16 + rlo) * K + kc * 32 + rhalf * 8;
            v8h blo = *(const v8h*)bp;
            v8h bhi = *(const v8h*)(bp + 16);
            v16h bm;
            #pragma unroll
            for (int i = 0; i < 8; ++i) { bm[i] = blo[i]; bm[8 + i] = bhi[i]; }
            acc[nt] = __builtin_amdgcn_wmma_f32_16x16x32_f16(
                false, a, false, bm, (short)0, acc[nt], false, false);
        }
    }
    #pragma unroll
    for (int nt = 0; nt < NT; ++nt) {
        float bv = bias[nt * 16 + rlo];
        #pragma unroll
        for (int r = 0; r < 8; ++r) {
            float x = acc[nt][r] + bv;
            actOut[(r + 8 * rhalf) * N + nt * 16 + rlo] = (_Float16)silu_f(x);
        }
    }
}

__global__ void __launch_bounds__(256, 1)
expr_kernel(const float* __restrict__ verts, const _Float16* __restrict__ w16,
            const float* __restrict__ ebias,
            const float* __restrict__ eb1, const float* __restrict__ eb2,
            const float* __restrict__ eb3, const float* __restrict__ eb4,
            const float* __restrict__ eb5,
            float* __restrict__ expr, int V)
{
    extern __shared__ __align__(16) _Float16 smem[];   // 64K halves W + 2 x 32K halves act
    const int tid = threadIdx.x, wave = tid >> 5, lane = tid & 31;
    const int b = blockIdx.y;
    const int rowBase = blockIdx.x * 128 + wave * 16;
    _Float16* wlds = smem;
    _Float16* myA  = smem + 65536 + wave * 4096;            // [16][256] f16
    _Float16* myB  = smem + 65536 + 32768 + wave * 4096;    // [16][256] f16
    float* gOut = expr + (size_t)b * V * 3;

    stage_w(w16 + OFF_W0A, wlds, 24576, tid);
    expr_layer0<256>(verts, V, wlds, ebias + b * 256, myA, lane, rowBase);
    stage_w(w16 + OFF_W1, wlds, 65536, tid);
    mlp_layer<256, 256, false>(myA, wlds, eb1, myB, nullptr, lane, rowBase, V);
    stage_w(w16 + OFF_W2, wlds, 32768, tid);
    mlp_layer<256, 128, false>(myB, wlds, eb2, myA, nullptr, lane, rowBase, V);
    stage_w(w16 + OFF_W3, wlds, 16384, tid);
    mlp_layer<128, 128, false>(myA, wlds, eb3, myB, nullptr, lane, rowBase, V);
    stage_w(w16 + OFF_W4, wlds, 16384, tid);
    mlp_layer<128, 128, false>(myB, wlds, eb4, myA, nullptr, lane, rowBase, V);
    stage_w(w16 + OFF_W5, wlds, 2048, tid);
    mlp_layer<128, 16, true>(myA, wlds, eb5, nullptr, gOut, lane, rowBase, V);
}

// =====================================================================
// K3: pose MLP (scalar) + rotation + final transform
// =====================================================================
__global__ void final_kernel(const float* __restrict__ verts, const float* __restrict__ feats,
                             const float* __restrict__ pw0,
                             const float* __restrict__ pw1, const float* __restrict__ pb1,
                             const float* __restrict__ pw2, const float* __restrict__ pb2,
                             const float* __restrict__ pw3, const float* __restrict__ pb3,
                             const float* __restrict__ pbias,
                             const float* __restrict__ expr, const float* __restrict__ tmpl,
                             const float* __restrict__ torigin, const float* __restrict__ scale,
                             float* __restrict__ out, int V)
{
    int v = blockIdx.x * 256 + threadIdx.x;
    int b = blockIdx.y;
    if (v >= V) return;

    float x0 = verts[v * 6 + 0], x1 = verts[v * 6 + 1], x2 = verts[v * 6 + 2];
    float e[21];
    e[0] = x0; e[1] = x1; e[2] = x2;
    #pragma unroll
    for (int n = 0; n < 3; ++n) {
        float s = (float)(1 << n);
        e[3 + 6 * n + 0] = __sinf(x0 * s); e[3 + 6 * n + 1] = __sinf(x1 * s); e[3 + 6 * n + 2] = __sinf(x2 * s);
        e[3 + 6 * n + 3] = __cosf(x0 * s); e[3 + 6 * n + 4] = __cosf(x1 * s); e[3 + 6 * n + 5] = __cosf(x2 * s);
    }
    // pose MLP: 21(+bias-folded 6) -> 64 -> 32 -> 32 -> 1, sigmoid
    float p0[64], p1[32], p2[32], po[1];
    dense<21, 64, 27, true >(pw0, pbias + b * 64, e,  p0);   // pw0 rows stride 27, cols 0..20
    dense<64, 32, 64, true >(pw1, pb1, p0, p1);
    dense<32, 32, 32, true >(pw2, pb2, p1, p2);
    dense<32,  1, 32, false>(pw3, pb3, p2, po);
    float w = sigm_f(po[0]);

    float a0 = feats[b * 62 + 53] * w;
    float a1 = feats[b * 62 + 54] * w;
    float a2 = feats[b * 62 + 55] * w;
    float cx = __cosf(a0), sx = __sinf(a0);
    float cy = __cosf(a1), sy = __sinf(a1);
    float cz = __cosf(a2), sz = __sinf(a2);
    // R = Rx @ Ry @ Rz (XYZ convention)
    float r00 = cy * cz,                 r01 = -cy * sz,                 r02 = sy;
    float r10 = sx * sy * cz + cx * sz,  r11 = -sx * sy * sz + cx * cz,  r12 = -sx * cy;
    float r20 = -cx * sy * cz + sx * sz, r21 = cx * sy * sz + sx * cz,   r22 = cx * cy;

    float ox = torigin[0] * 0.1f, oy = torigin[1] * 0.1f, oz = torigin[2] * 0.1f;
    float sc = scale[0];
    size_t bi = ((size_t)b * V + v) * 3;
    float e0 = verts[v * 6 + 0] + expr[bi + 0] + tmpl[v * 3 + 0];
    float e1 = verts[v * 6 + 1] + expr[bi + 1] + tmpl[v * 3 + 1];
    float e2 = verts[v * 6 + 2] + expr[bi + 2] + tmpl[v * 3 + 2];
    float l0 = (e0 - ox) * sc, l1 = (e1 - oy) * sc, l2 = (e2 - oz) * sc;

    out[bi + 0] = l0 * r00 + l1 * r10 + l2 * r20 + feats[b * 62 + 56] + ox;
    out[bi + 1] = l0 * r01 + l1 * r11 + l2 * r21 + feats[b * 62 + 57] + oy;
    out[bi + 2] = l0 * r02 + l1 * r12 + l2 * r22 + feats[b * 62 + 58] + oz;
}

// =====================================================================
extern "C" void kernel_launch(void* const* d_in, const int* in_sizes, int n_in,
                              void* d_out, int out_size, void* d_ws, size_t ws_size,
                              hipStream_t stream)
{
    const float* verts = (const float*)d_in[0];
    const float* feats = (const float*)d_in[1];
    const float* ew0 = (const float*)d_in[2];  const float* eb0 = (const float*)d_in[3];
    const float* ew1 = (const float*)d_in[4];  const float* eb1 = (const float*)d_in[5];
    const float* ew2 = (const float*)d_in[6];  const float* eb2 = (const float*)d_in[7];
    const float* ew3 = (const float*)d_in[8];  const float* eb3 = (const float*)d_in[9];
    const float* ew4 = (const float*)d_in[10]; const float* eb4 = (const float*)d_in[11];
    const float* ew5 = (const float*)d_in[12]; const float* eb5 = (const float*)d_in[13];
    const float* tw0 = (const float*)d_in[14]; const float* tb0 = (const float*)d_in[15];
    const float* tw1 = (const float*)d_in[16]; const float* tb1 = (const float*)d_in[17];
    const float* tw2 = (const float*)d_in[18]; const float* tb2 = (const float*)d_in[19];
    const float* tw3 = (const float*)d_in[20]; const float* tb3 = (const float*)d_in[21];
    const float* pw0 = (const float*)d_in[22]; const float* pb0 = (const float*)d_in[23];
    const float* pw1 = (const float*)d_in[24]; const float* pb1 = (const float*)d_in[25];
    const float* pw2 = (const float*)d_in[26]; const float* pb2 = (const float*)d_in[27];
    const float* pw3 = (const float*)d_in[28]; const float* pb3 = (const float*)d_in[29];
    const float* torg = (const float*)d_in[30];
    const float* scal = (const float*)d_in[31];

    const int V = in_sizes[0] / 6;
    const int B = in_sizes[1] / 62;

    char* ws = (char*)d_ws;
    size_t off_w16   = 0;
    size_t off_expr  = off_w16 + (size_t)W16_TOTAL * 2;           // 315392 B (16B aligned)
    size_t off_tmpl  = off_expr + (size_t)B * V * 3 * 4;
    size_t off_ebias = off_tmpl + (size_t)V * 3 * 4;
    size_t off_pbias = off_ebias + (size_t)B * 256 * 4;

    _Float16* w16  = (_Float16*)(ws + off_w16);
    float* exprbuf = (float*)(ws + off_expr);
    float* tmplbuf = (float*)(ws + off_tmpl);
    float* ebias   = (float*)(ws + off_ebias);
    float* pbias   = (float*)(ws + off_pbias);

    pack_weights<<<(W16_TOTAL + 255) / 256, 256, 0, stream>>>(ew0, ew1, ew2, ew3, ew4, ew5, w16);
    bias_kernel<<<B, 256, 0, stream>>>(feats, ew0, eb0, pw0, pb0, ebias, pbias);
    tmpl_kernel<<<(V + 255) / 256, 256, 0, stream>>>(verts, tw0, tb0, tw1, tb1, tw2, tb2, tw3, tb3,
                                                     tmplbuf, V);
    dim3 g2((V + 127) / 128, B);
    expr_kernel<<<g2, 256, 262144, stream>>>(verts, w16, ebias, eb1, eb2, eb3, eb4, eb5, exprbuf, V);
    dim3 g3((V + 255) / 256, B);
    final_kernel<<<g3, 256, 0, stream>>>(verts, feats, pw0, pw1, pb1, pw2, pb2, pw3, pb3,
                                         pbias, exprbuf, tmplbuf, torg, scal, (float*)d_out, V);
}
